// SelfAttentionModule_85177791415014
// MI455X (gfx1250) — matrix operand order repaired and checked by reading, not compiled, not run on hardware
//
#include <hip/hip_runtime.h>
#include <hip/hip_bf16.h>

// ---------------------------------------------------------------------------
// Local self-attention (K=7 window) for MI455X / gfx1250, wave32 + WMMA bf16.
//
// Pipeline (all on `stream`, workspace-resident, L2-friendly):
//   1. prep_wqkv : fold BN into Wq/Wk, stack [Wq*sq; Wk*sk; Wv] -> bf16 [256][256]
//   2. prep_ww   : Ww -> bf16 [256][128]
//   3. x_trans   : x [B,C,H,W] f32 -> xT [N][256] bf16 (LDS-tiled transpose)
//   4. gemm_qkv  : v_wmma_f32_16x16x32_bf16, D -> qkv f32 [N][256] (+bias)
//   5. attn      : per-pixel 49-way dot/softmax/weighted-sum -> ctx bf16 [N][128]
//   6. gemm_out  : WMMA, D -> out [B,256,H,W] f32 (+bw)
// ---------------------------------------------------------------------------

typedef __attribute__((ext_vector_type(16))) __bf16 bf16x16;
typedef __attribute__((ext_vector_type(8)))  float  f32x8;

#define BATCH 2
#define HW    9216          // 96*96
#define NPIX  18432         // BATCH*HW
#define CIN   256
#define KC    64
#define VC    128
#define OC    256

__device__ __forceinline__ unsigned short f2bf(float f) {
    unsigned int u = __float_as_uint(f);
    u += 0x7FFFu + ((u >> 16) & 1u);   // round-to-nearest-even
    return (unsigned short)(u >> 16);
}

union FragU { bf16x16 v; uint4 q[2]; };

// A-matrix 16x32 bf16 fragment (ISA 7.12.2): lanes 0-15 row M=l, K={k0..k0+7, k0+16..k0+23};
// lanes 16-31 same rows, K shifted by +8. Two 16B loads per lane.
__device__ __forceinline__ bf16x16 load_a_frag(const unsigned short* __restrict__ A,
                                               int lda, int m0, int k0, int lane) {
    int m  = m0 + (lane & 15);
    int kb = k0 + ((lane & 16) >> 1);          // +8 ushorts for hi half-wave
    const unsigned short* p = A + m * lda + kb;
    FragU u;
    u.q[0] = *(const uint4*)(p);               // K = kb .. kb+7
    u.q[1] = *(const uint4*)(p + 16);          // K = kb+16 .. kb+23
    return u.v;
}

// B-matrix 32x16 bf16 fragment: lane l holds column N=l&15; lanes 0-15 K={k0..k0+7, k0+16..k0+23},
// lanes 16-31 same, K shifted by +8 (matches A fragment).
// Two 16B loads per lane from pixel-major X^T [N][Ck].
__device__ __forceinline__ bf16x16 load_b_frag(const unsigned short* __restrict__ B,
                                               int ldb, int n0, int k0, int lane) {
    int n  = n0 + (lane & 15);
    int kb = k0 + ((lane & 16) >> 1);          // +8 ushorts for hi half-wave
    const unsigned short* p = B + n * ldb + kb;
    FragU u;
    u.q[0] = *(const uint4*)(p);
    u.q[1] = *(const uint4*)(p + 16);
    return u.v;
}

// ---------------------------------------------------------------------------
// 1. Build stacked effective QKV weights (BN folded) + bias.
//    rows 0..63 = wq*sq, 64..127 = wk*sk, 128..255 = wv
__global__ void prep_wqkv(const float* __restrict__ wq, const float* __restrict__ bq,
                          const float* __restrict__ gq, const float* __restrict__ betaq,
                          const float* __restrict__ mq, const float* __restrict__ vq,
                          const float* __restrict__ wk, const float* __restrict__ bk,
                          const float* __restrict__ gk, const float* __restrict__ betak,
                          const float* __restrict__ mk, const float* __restrict__ vk,
                          const float* __restrict__ wv, const float* __restrict__ bv,
                          unsigned short* __restrict__ Wqkv, float* __restrict__ bias) {
    int o = blockIdx.x;        // 0..255
    int c = threadIdx.x;       // 0..255
    float w, bb;
    if (o < KC) {
        float s = gq[o] * rsqrtf(vq[o] + 1e-5f);
        w  = wq[o * CIN + c] * s;
        bb = (bq[o] - mq[o]) * s + betaq[o];
    } else if (o < 2 * KC) {
        int oo = o - KC;
        float s = gk[oo] * rsqrtf(vk[oo] + 1e-5f);
        w  = wk[oo * CIN + c] * s;
        bb = (bk[oo] - mk[oo]) * s + betak[oo];
    } else {
        int oo = o - 2 * KC;
        w  = wv[oo * CIN + c];
        bb = bv[oo];
    }
    Wqkv[o * CIN + c] = f2bf(w);
    if (c == 0) bias[o] = bb;
}

// 2. Ww -> bf16
__global__ void prep_ww(const float* __restrict__ ww, unsigned short* __restrict__ W) {
    int o = blockIdx.x;        // 0..255
    int c = threadIdx.x;       // 0..127
    W[o * VC + c] = f2bf(ww[o * VC + c]);
}

// ---------------------------------------------------------------------------
// 3. x [B,C,HW] f32 -> xT [N][C] bf16, LDS-tiled 32x32 transpose.
__global__ void x_trans(const float* __restrict__ x, unsigned short* __restrict__ xT) {
    __shared__ unsigned short tile[32][33];
    int b  = blockIdx.z;
    int c0 = blockIdx.y * 32;
    int s0 = blockIdx.x * 32;
    int tx = threadIdx.x, ty = threadIdx.y;    // (32, 8)
#pragma unroll
    for (int i = 0; i < 4; ++i) {
        int c = c0 + ty + 8 * i;
        tile[ty + 8 * i][tx] = f2bf(x[((size_t)(b * CIN + c)) * HW + s0 + tx]);
    }
    __syncthreads();
#pragma unroll
    for (int i = 0; i < 4; ++i) {
        int s = s0 + ty + 8 * i;
        xT[((size_t)(b * HW + s)) * CIN + c0 + tx] = tile[tx][ty + 8 * i];
    }
}

// ---------------------------------------------------------------------------
// 4. QKV GEMM: D[256 x N] = Wqkv[256 x 256] * xT^T, stored pixel-major [N][256]+bias.
//    Per-wave: one 16-M tile x four 16-N tiles, 8 K-steps of 32.
__global__ void __launch_bounds__(256) gemm_qkv(const unsigned short* __restrict__ Wqkv,
                                                const unsigned short* __restrict__ xT,
                                                const float* __restrict__ bias,
                                                float* __restrict__ qkv) {
    const int lane = threadIdx.x & 31;
    const int wid  = blockIdx.x * (blockDim.x >> 5) + (threadIdx.x >> 5);
    const int mt   = wid & 15;                 // 16 M-tiles (256 channels)
    const int ng   = wid >> 4;                 // 288 N-groups of 4 tiles
    const int m0   = mt * 16;

    f32x8 acc[4] = {};
    for (int k0 = 0; k0 < CIN; k0 += 32) {
        bf16x16 a = load_a_frag(Wqkv, CIN, m0, k0, lane);
#pragma unroll
        for (int j = 0; j < 4; ++j) {
            bf16x16 b = load_b_frag(xT, CIN, (ng * 4 + j) * 16, k0, lane);
            acc[j] = __builtin_amdgcn_wmma_f32_16x16x32_bf16(
                         false, a, false, b, (short)0, acc[j], false, false);
        }
    }
    // D layout: lane holds N=lane&15, M = (0..7 | 8..15) contiguous -> 32B store
    const int n_l = lane & 15;
    const int m_l = m0 + ((lane & 16) >> 1);
#pragma unroll
    for (int j = 0; j < 4; ++j) {
        int n = (ng * 4 + j) * 16 + n_l;
        float* p = qkv + (size_t)n * OC + m_l;
#pragma unroll
        for (int r = 0; r < 8; ++r) p[r] = acc[j][r] + bias[m_l + r];
    }
}

// ---------------------------------------------------------------------------
// 5. Attention: per-thread pixel. q=qkv[n][0:64], k=qkv[nn][64:128], v=qkv[nn][128:256].
//    Zero-padded neighbors contribute sim=0 (matches reference unfold-pad semantics)
//    and v=0. ctx written as bf16 [N][128].
__global__ void __launch_bounds__(256) attn(const float* __restrict__ qkv,
                                            unsigned short* __restrict__ ctx) {
    int n  = blockIdx.x * blockDim.x + threadIdx.x;   // 0..NPIX-1 (exact)
    int b  = n / HW;
    int hw = n - b * HW;
    int h  = hw / 96, w = hw - h * 96;

    float qv[KC];
    const float* q = qkv + (size_t)n * OC;
#pragma unroll
    for (int c = 0; c < KC; ++c) qv[c] = q[c];

    float sim[49];
#pragma unroll
    for (int p = 0; p < 49; ++p) {
        int hh = h + p / 7 - 3;
        int wp = w + p % 7 - 3;
        float s = 0.f;
        if ((unsigned)hh < 96u && (unsigned)wp < 96u) {
            const float* kp = qkv + (size_t)(b * HW + hh * 96 + wp) * OC + KC;
#pragma unroll
            for (int c = 0; c < KC; ++c) s = fmaf(qv[c], kp[c], s);
        }
        sim[p] = s;
    }

    float mx = sim[0];
#pragma unroll
    for (int p = 1; p < 49; ++p) mx = fmaxf(mx, sim[p]);
    float ssum = 0.f;
#pragma unroll
    for (int p = 0; p < 49; ++p) { sim[p] = __expf(sim[p] - mx); ssum += sim[p]; }
    float inv = 1.f / ssum;

    unsigned short* cp = ctx + (size_t)n * VC;
#pragma unroll
    for (int c0 = 0; c0 < VC; c0 += 16) {
        float acc[16];
#pragma unroll
        for (int j = 0; j < 16; ++j) acc[j] = 0.f;
#pragma unroll
        for (int p = 0; p < 49; ++p) {
            int hh = h + p / 7 - 3;
            int wp = w + p % 7 - 3;
            if ((unsigned)hh < 96u && (unsigned)wp < 96u) {
                const float* vp = qkv + (size_t)(b * HW + hh * 96 + wp) * OC + 2 * KC + c0;
#pragma unroll
                for (int j = 0; j < 16; ++j) acc[j] = fmaf(sim[p], vp[j], acc[j]);
            }
        }
        unsigned int pk[8];
#pragma unroll
        for (int j = 0; j < 8; ++j) {
            pk[j] = (unsigned int)f2bf(acc[2 * j] * inv)
                  | ((unsigned int)f2bf(acc[2 * j + 1] * inv) << 16);
        }
        *(uint4*)(cp + c0)     = make_uint4(pk[0], pk[1], pk[2], pk[3]);
        *(uint4*)(cp + c0 + 8) = make_uint4(pk[4], pk[5], pk[6], pk[7]);
    }
}

// ---------------------------------------------------------------------------
// 6. Output GEMM: out[B,256,HW] = Ww[256 x 128] * ctx^T + bw, channel-major store.
__global__ void __launch_bounds__(256) gemm_out(const unsigned short* __restrict__ Ww,
                                                const unsigned short* __restrict__ ctx,
                                                const float* __restrict__ bw,
                                                float* __restrict__ out) {
    const int lane = threadIdx.x & 31;
    const int wid  = blockIdx.x * (blockDim.x >> 5) + (threadIdx.x >> 5);
    const int mt   = wid & 15;                 // 16 M-tiles (256 out channels)
    const int ng   = wid >> 4;                 // 288 N-groups
    const int m0   = mt * 16;

    f32x8 acc[4] = {};
    for (int k0 = 0; k0 < VC; k0 += 32) {
        bf16x16 a = load_a_frag(Ww, VC, m0, k0, lane);
#pragma unroll
        for (int j = 0; j < 4; ++j) {
            bf16x16 b = load_b_frag(ctx, VC, (ng * 4 + j) * 16, k0, lane);
            acc[j] = __builtin_amdgcn_wmma_f32_16x16x32_bf16(
                         false, a, false, b, (short)0, acc[j], false, false);
        }
    }
    const int n_l = lane & 15;
    const int m_l = m0 + ((lane & 16) >> 1);
#pragma unroll
    for (int j = 0; j < 4; ++j) {
        int n  = (ng * 4 + j) * 16 + n_l;
        int b  = n / HW;                       // tiles never cross batch (16 | 9216)
        int hw = n - b * HW;
        float* p = out + (size_t)b * OC * HW + (size_t)m_l * HW + hw;
#pragma unroll
        for (int r = 0; r < 8; ++r) p[(size_t)r * HW] = acc[j][r] + bw[m_l + r];
    }
}

// ---------------------------------------------------------------------------
extern "C" void kernel_launch(void* const* d_in, const int* in_sizes, int n_in,
                              void* d_out, int out_size, void* d_ws, size_t ws_size,
                              hipStream_t stream) {
    const float* x     = (const float*)d_in[0];
    const float* wq    = (const float*)d_in[1];
    const float* bq    = (const float*)d_in[2];
    const float* gq    = (const float*)d_in[3];
    const float* betaq = (const float*)d_in[4];
    const float* mq    = (const float*)d_in[5];
    const float* vq    = (const float*)d_in[6];
    const float* wk    = (const float*)d_in[7];
    const float* bk    = (const float*)d_in[8];
    const float* gk    = (const float*)d_in[9];
    const float* betak = (const float*)d_in[10];
    const float* mk    = (const float*)d_in[11];
    const float* vk    = (const float*)d_in[12];
    const float* wv    = (const float*)d_in[13];
    const float* bv    = (const float*)d_in[14];
    const float* ww    = (const float*)d_in[15];
    const float* bw    = (const float*)d_in[16];
    float* out = (float*)d_out;

    char* ws = (char*)d_ws;
    unsigned short* Wqkv = (unsigned short*)(ws + 0);          // 256*256*2  = 131072
    float*          bias = (float*)(ws + 131072);              // 256*4      = 1024
    unsigned short* Wout = (unsigned short*)(ws + 132096);     // 256*128*2  = 65536
    unsigned short* xT   = (unsigned short*)(ws + 197632);     // 18432*256*2= 9437184
    float*          qkv  = (float*)(ws + 9634816);             // 18432*256*4= 18874368
    unsigned short* ctx  = (unsigned short*)(ws + 28509184);   // 18432*128*2= 4718592
                                                               // total 33227776 B

    prep_wqkv<<<OC, CIN, 0, stream>>>(wq, bq, gq, betaq, mq, vq,
                                      wk, bk, gk, betak, mk, vk,
                                      wv, bv, Wqkv, bias);
    prep_ww<<<OC, VC, 0, stream>>>(ww, Wout);
    x_trans<<<dim3(HW / 32, CIN / 32, BATCH), dim3(32, 8), 0, stream>>>(x, xT);
    gemm_qkv<<<576, 256, 0, stream>>>(Wqkv, xT, bias, qkv);    // 16 M * 288 N-grp / 8 waves
    attn<<<NPIX / 256, 256, 0, stream>>>(qkv, ctx);
    gemm_out<<<576, 256, 0, stream>>>(Wout, ctx, bw, out);
}
